// GraphSAGEPredictor_18262200942971
// MI455X (gfx1250) — matrix operand order repaired and checked by reading, not compiled, not run on hardware
//
#include <hip/hip_runtime.h>
#include <hip/hip_bf16.h>
#include <cmath>

// ---------- types for CDNA5 WMMA ----------
typedef __bf16 bf16x16 __attribute__((ext_vector_type(16)));
typedef __bf16 bf16x8  __attribute__((ext_vector_type(8)));
typedef float  f32x8   __attribute__((ext_vector_type(8)));

#define HIDC 128

// ---------- order-preserving float <-> u32 keys for atomic max ----------
__device__ __forceinline__ unsigned enc_key(float f) {
    unsigned u = __float_as_uint(f);
    return (u & 0x80000000u) ? ~u : (u | 0x80000000u);
}
__device__ __forceinline__ float dec_key(unsigned k) {
    unsigned u = (k & 0x80000000u) ? (k ^ 0x80000000u) : ~k;
    // never-written (k==0 -> NaN) and -inf both map to 0, matching
    // jnp.where(isfinite(segment_max), agg, 0)
    if ((u & 0x7F800000u) == 0x7F800000u) return 0.0f;
    return __uint_as_float(u);
}

// ---------- WMMA fragment loaders (contiguous k = 16*(lane/16) + i order, wave32) ----------
// A (16x32 bf16): lanes 0-15 = rows M=0..15 with K 0..15;
//                 lanes 16-31 = same rows with K 16..31
__device__ __forceinline__ bf16x16 load_fragA_f32(const float* M, int ld,
                                                  int row0, int k0, int lane) {
    const float* p = M + (size_t)(row0 + (lane & 15)) * ld + k0 + ((lane >> 4) << 4);
    bf16x16 a;
#pragma unroll
    for (int i = 0; i < 8; ++i) a[i] = (__bf16)p[i];
#pragma unroll
    for (int i = 0; i < 8; ++i) a[8 + i] = (__bf16)p[8 + i];
    return a;
}
__device__ __forceinline__ bf16x16 load_fragA_keys(const unsigned* M, int ld,
                                                   int row0, int k0, int lane) {
    const unsigned* p = M + (size_t)(row0 + (lane & 15)) * ld + k0 + ((lane >> 4) << 4);
    bf16x16 a;
#pragma unroll
    for (int i = 0; i < 8; ++i) a[i] = (__bf16)dec_key(p[i]);
#pragma unroll
    for (int i = 0; i < 8; ++i) a[8 + i] = (__bf16)dec_key(p[8 + i]);
    return a;
}
// B (32x16 bf16): B[k][n] = W[n0+n][k0+k], W pre-converted bf16 row-major [nOut x ld].
// lanes 0-15 = col n, K=0..15 ; lanes 16-31 = col n, K=16..31 (16 contiguous bf16 = 2 x b128)
__device__ __forceinline__ bf16x16 load_fragB(const __bf16* W, int ld,
                                              int n0, int k0, int lane) {
    const __bf16* p = W + (size_t)(n0 + (lane & 15)) * ld + k0 + ((lane >> 4) << 4);
    bf16x8 lo = *(const bf16x8*)p;
    bf16x8 hi = *(const bf16x8*)(p + 8);
    bf16x16 b;
#pragma unroll
    for (int i = 0; i < 8; ++i) { b[i] = lo[i]; b[8 + i] = hi[i]; }
    return b;
}
__device__ __forceinline__ f32x8 wmma_bf16(bf16x16 a, bf16x16 b, f32x8 c) {
    return __builtin_amdgcn_wmma_f32_16x16x32_bf16(false, a, false, b,
                                                   (short)0, c, false, false);
}
// butterfly sum over the 16-lane half-groups of the C/D fragment layout
__device__ __forceinline__ float hsum16(float v) {
#pragma unroll
    for (int m = 1; m < 16; m <<= 1) v += __shfl_xor(v, m, 16);
    return v;
}

// ---------- one-shot weight f32 -> bf16 conversion (loop-invariant) ----------
__global__ void cvt_weights_kernel(const float* __restrict__ in,
                                   __bf16* __restrict__ out, int n) {
    int i = blockIdx.x * blockDim.x + threadIdx.x;
    if (i < n) out[i] = (__bf16)in[i];
}

// ---------- edge scatter-max (element-parallel atomics into L2) ----------
__global__ void scatter_max_kernel(const long long* __restrict__ src,
                                   const long long* __restrict__ dst,
                                   const float* __restrict__ feat,
                                   unsigned* __restrict__ keys,
                                   int C, int c4log2, long long total) {
    long long tid = (long long)blockIdx.x * blockDim.x + threadIdx.x;
    if (tid >= total) return;
    long long e = tid >> c4log2;
    int g = ((int)tid & ((1 << c4log2) - 1)) << 2;
    long long s = src[e], d = dst[e];
    const float4 v = *(const float4*)(feat + (size_t)s * C + g);
    unsigned* kp = keys + (size_t)d * C + g;
    atomicMax(kp + 0, enc_key(v.x));
    atomicMax(kp + 1, enc_key(v.y));
    atomicMax(kp + 2, enc_key(v.z));
    atomicMax(kp + 3, enc_key(v.w));
}

// ---------- fused SAGE layer: agg@Wl^T + x@Wr^T + bl -> LayerNorm -> ReLU ----------
template <int KDIM>
__global__ void __launch_bounds__(256)
sage_layer_kernel(const unsigned* __restrict__ aggK, const float* __restrict__ xr,
                  const __bf16* __restrict__ Wl, const float* __restrict__ bl,
                  const __bf16* __restrict__ Wr,
                  const float* __restrict__ gamma, const float* __restrict__ beta,
                  float* __restrict__ out, int nrows) {
    const int lane = threadIdx.x & 31;
    const int wave = threadIdx.x >> 5;
    const int tiles = nrows >> 4;                 // nrows % 16 == 0 (N = 100000)
    int tile = blockIdx.x * 8 + wave;
    if (tile >= tiles) tile = tiles - 1;          // duplicate (identical) work: EXEC stays all-ones
    const int row0 = tile << 4;

    f32x8 acc[8];
#pragma unroll
    for (int t = 0; t < 8; ++t) acc[t] = (f32x8){};

#pragma unroll
    for (int k0 = 0; k0 < KDIM; k0 += 32) {
        bf16x16 aA = load_fragA_keys(aggK, KDIM, row0, k0, lane);
        bf16x16 aR = load_fragA_f32(xr, KDIM, row0, k0, lane);
#pragma unroll
        for (int t = 0; t < 8; ++t) {
            acc[t] = wmma_bf16(aA, load_fragB(Wl, KDIM, t * 16, k0, lane), acc[t]);
            acc[t] = wmma_bf16(aR, load_fragB(Wr, KDIM, t * 16, k0, lane), acc[t]);
        }
    }

    const int nlo = lane & 15;
    float blv[8], gv[8], bev[8];
#pragma unroll
    for (int t = 0; t < 8; ++t) {
        int n = t * 16 + nlo;
        blv[t] = bl[n]; gv[t] = gamma[n]; bev[t] = beta[n];
    }
    // bias (neighbor path) before LN stats
#pragma unroll
    for (int t = 0; t < 8; ++t)
#pragma unroll
        for (int v = 0; v < 8; ++v) acc[t][v] += blv[t];

    // per-row LayerNorm + ReLU; row of element (t,v) is row0 + v + 8*(lane>>4)
#pragma unroll
    for (int v = 0; v < 8; ++v) {
        float s = 0.f, q = 0.f;
#pragma unroll
        for (int t = 0; t < 8; ++t) { float x = acc[t][v]; s += x; q += x * x; }
        s = hsum16(s); q = hsum16(q);
        float mu = s * (1.0f / HIDC);
        float var = q * (1.0f / HIDC) - mu * mu;
        float rs = rsqrtf(var + 1e-5f);
        int row = row0 + v + ((lane >> 4) << 3);
        float* op = out + (size_t)row * HIDC + nlo;
#pragma unroll
        for (int t = 0; t < 8; ++t) {
            float y = (acc[t][v] - mu) * rs * gv[t] + bev[t];
            op[t * 16] = fmaxf(y, 0.0f);
        }
    }
}

// ---------- fused MLP head: relu(h@W1^T+b1) @ W2^T + b2 -> sigmoid ----------
__global__ void __launch_bounds__(256)
head_kernel(const float* __restrict__ h, const __bf16* __restrict__ W1,
            const float* __restrict__ b1, const float* __restrict__ W2,
            const float* __restrict__ b2, float* __restrict__ out, int nrows) {
    const int lane = threadIdx.x & 31;
    const int wave = threadIdx.x >> 5;
    const int tiles = nrows >> 4;
    int tile = blockIdx.x * 8 + wave;
    if (tile >= tiles) tile = tiles - 1;
    const int row0 = tile << 4;

    f32x8 acc[4];
#pragma unroll
    for (int t = 0; t < 4; ++t) acc[t] = (f32x8){};

#pragma unroll
    for (int k0 = 0; k0 < HIDC; k0 += 32) {
        bf16x16 a = load_fragA_f32(h, HIDC, row0, k0, lane);
#pragma unroll
        for (int t = 0; t < 4; ++t)
            acc[t] = wmma_bf16(a, load_fragB(W1, HIDC, t * 16, k0, lane), acc[t]);
    }

    const int nlo = lane & 15;
    float b1v[4], w2v[4];
#pragma unroll
    for (int t = 0; t < 4; ++t) { b1v[t] = b1[t * 16 + nlo]; w2v[t] = W2[t * 16 + nlo]; }
    const float bb = b2[0];

#pragma unroll
    for (int v = 0; v < 8; ++v) {
        float r = 0.f;
#pragma unroll
        for (int t = 0; t < 4; ++t) r += fmaxf(acc[t][v] + b1v[t], 0.0f) * w2v[t];
        r = hsum16(r);
        if (nlo == 0) {
            int row = row0 + v + ((lane >> 4) << 3);
            out[row] = 1.0f / (1.0f + __expf(-(r + bb)));
        }
    }
}

extern "C" void kernel_launch(void* const* d_in, const int* in_sizes, int n_in,
                              void* d_out, int out_size, void* d_ws, size_t ws_size,
                              hipStream_t stream) {
    const float*     x   = (const float*)d_in[0];
    const long long* ei  = (const long long*)d_in[1];
    const float* Wl0 = (const float*)d_in[2];
    const float* bl0 = (const float*)d_in[3];
    const float* Wr0 = (const float*)d_in[4];
    const float* g0  = (const float*)d_in[5];
    const float* be0 = (const float*)d_in[6];
    const float* Wl1 = (const float*)d_in[7];
    const float* bl1 = (const float*)d_in[8];
    const float* Wr1 = (const float*)d_in[9];
    const float* g1  = (const float*)d_in[10];
    const float* be1 = (const float*)d_in[11];
    const float* W1  = (const float*)d_in[12];
    const float* b1  = (const float*)d_in[13];
    const float* W2  = (const float*)d_in[14];
    const float* b2  = (const float*)d_in[15];
    float* out = (float*)d_out;

    const int       N   = in_sizes[0] / 64;
    const long long E   = in_sizes[1] / 2;
    const long long* src  = ei;
    const long long* dstp = ei + E;

    // ---- workspace layout ----
    char* ws = (char*)d_ws;
    unsigned* keys0 = (unsigned*)(ws);
    float*    h1    = (float*)   (ws + (size_t)N * 64 * 4);
    unsigned* keys1 = (unsigned*)(ws + (size_t)N * (64 + 128) * 4);
    float*    h2    = (float*)   (ws + (size_t)N * (64 + 128 + 128) * 4);
    __bf16*   wb    = (__bf16*)  (ws + (size_t)N * (64 + 128 + 128 + 128) * 4);
    __bf16* Wl0b = wb;                 // 128*64
    __bf16* Wr0b = wb + 8192;          // 128*64
    __bf16* Wl1b = wb + 16384;         // 128*128
    __bf16* Wr1b = wb + 32768;         // 128*128
    __bf16* W1b  = wb + 49152;         // 64*128

    // ---- one-shot weight conversion to bf16 (loop-invariant, ~230 KB total) ----
    cvt_weights_kernel<<<(8192  + 255) / 256, 256, 0, stream>>>(Wl0, Wl0b, 8192);
    cvt_weights_kernel<<<(8192  + 255) / 256, 256, 0, stream>>>(Wr0, Wr0b, 8192);
    cvt_weights_kernel<<<(16384 + 255) / 256, 256, 0, stream>>>(Wl1, Wl1b, 16384);
    cvt_weights_kernel<<<(16384 + 255) / 256, 256, 0, stream>>>(Wr1, Wr1b, 16384);
    cvt_weights_kernel<<<(8192  + 255) / 256, 256, 0, stream>>>(W1,  W1b,  8192);

    const int tiles   = N / 16;
    const int gblocks = (tiles + 7) / 8;

    // ---- layer 0 ----
    hipMemsetAsync(keys0, 0, (size_t)N * 64 * sizeof(unsigned), stream);
    {
        long long total = E * 16;                  // E * (64/4)
        int blocks = (int)((total + 255) / 256);
        scatter_max_kernel<<<blocks, 256, 0, stream>>>(src, dstp, x, keys0, 64, 4, total);
    }
    sage_layer_kernel<64><<<gblocks, 256, 0, stream>>>(keys0, x, Wl0b, bl0, Wr0b, g0, be0, h1, N);

    // ---- layer 1 ----
    hipMemsetAsync(keys1, 0, (size_t)N * 128 * sizeof(unsigned), stream);
    {
        long long total = E * 32;                  // E * (128/4)
        int blocks = (int)((total + 255) / 256);
        scatter_max_kernel<<<blocks, 256, 0, stream>>>(src, dstp, h1, keys1, 128, 5, total);
    }
    sage_layer_kernel<128><<<gblocks, 256, 0, stream>>>(keys1, h1, Wl1b, bl1, Wr1b, g1, be1, h2, N);

    // ---- MLP head ----
    head_kernel<<<gblocks, 256, 0, stream>>>(h2, W1b, b1, W2, b2, out, N);
}
